// knrm_35931696398610
// MI455X (gfx1250) — matrix operand order repaired and checked by reading, not compiled, not run on hardware
//
#include <hip/hip_runtime.h>

// K-NRM on MI455X (gfx1250, wave32, WMMA).
// Stage 1: gather + L2-normalize embeddings -> f16 (padded E 300 -> 320).
// Stage 2: per-batch sim GEMM via v_wmma_f32_16x16x32_f16, fused Gaussian
//          kernel pooling using a geometric recurrence (3 exps per sim
//          instead of 21), packed-f32 row pairs, LDS reduce, log + dense.

#define B_  128
#define Q_  32
#define D_  512
#define E_  300
#define EP_ 320   // E padded to multiple of 32 (10 WMMA k-steps)
#define K_  21

typedef _Float16 v16h __attribute__((ext_vector_type(16)));
typedef _Float16 v8h  __attribute__((ext_vector_type(8)));
typedef float    v8f  __attribute__((ext_vector_type(8)));
typedef float    v2f  __attribute__((ext_vector_type(2)));

static __device__ __forceinline__ v16h cat16(v8h lo, v8h hi) {
  return __builtin_shufflevector(lo, hi, 0,1,2,3,4,5,6,7,8,9,10,11,12,13,14,15);
}

// exp(-(s-mu_k)^2/(2*0.1^2)) recurrence constants: e^(10*mu_k - 0.5) for
// k = 1..19 with mu_k = 0.95 - 0.1(k-1)  ->  e^9, e^8, ..., e^-9.
static __device__ const float EK[19] = {
  8103.083927575384f,   2980.9579870417283f,  1096.6331584284585f,
  403.4287934927351f,   148.4131591025766f,   54.598150033144236f,
  20.085536923187668f,  7.38905609893065f,    2.718281828459045f,
  1.0f,                 0.36787944117144233f, 0.1353352832366127f,
  0.049787068367863944f,0.01831563888873418f, 0.006737946999085467f,
  0.0024787521766663585f,0.0009118819655545162f,0.00033546262790251185f,
  0.0001234098040866796f
};

#define CC0 (-721347.5204444817f)   // -log2(e)/(2*0.001^2)
#define CC1 (-72.13475204444817f)   // -log2(e)/(2*0.1^2)
#define RC  (-14.426950408889634f)  // -10*log2(e)

// ---------------- Stage 1: gather + normalize + f32->f16 ----------------
// One wave per token row. 69632 rows total; grid sized exactly.
__global__ void __launch_bounds__(256)
gather_norm_kernel(const int* __restrict__ qidx,
                   const int* __restrict__ didx,
                   const float* __restrict__ emb,
                   _Float16* __restrict__ qws,
                   _Float16* __restrict__ dws) {
  const int gw   = (blockIdx.x * blockDim.x + threadIdx.x) >> 5;
  const int lane = threadIdx.x & 31;

  int tok;
  _Float16* dst;
  if (gw < B_ * Q_) {
    tok = qidx[gw];
    dst = qws + (size_t)gw * EP_;
  } else {
    const int r = gw - B_ * Q_;
    tok = didx[r];
    dst = dws + (size_t)r * EP_;
  }
  const float* src = emb + (size_t)tok * E_;

  float x[10];
  float ss = 0.f;
#pragma unroll
  for (int i = 0; i < 10; ++i) {
    const int idx = lane + 32 * i;
    const float v = (idx < E_) ? src[idx] : 0.f;
    x[i] = v;
    ss += v * v;
  }
#pragma unroll
  for (int off = 16; off > 0; off >>= 1)
    ss += __shfl_xor(ss, off, 32);

  const float inv = 1.f / fmaxf(sqrtf(ss), 1e-12f);
#pragma unroll
  for (int i = 0; i < 10; ++i) {
    const int idx = lane + 32 * i;
    dst[idx] = (idx < E_) ? (_Float16)(x[i] * inv) : (_Float16)0.f;
  }
}

// ---------------- Stage 2: WMMA sim + fused kernel pooling ----------------
// One 256-thread workgroup (8 waves) per batch element.
// wave w: mtile = w&1 (16 query rows), ntiles = (w>>1) + 4*j, j=0..7.
__global__ void __launch_bounds__(256)
knrm_pool_kernel(const _Float16* __restrict__ qws,
                 const _Float16* __restrict__ dws,
                 const int* __restrict__ qlen_arr,
                 const int* __restrict__ dlen_arr,
                 const float* __restrict__ dense_w,
                 const float* __restrict__ dense_b,
                 float* __restrict__ out) {
  __shared__ float pool[Q_][K_];   // pooling_sum[b, q, k] for this b
  __shared__ float lps_sh[K_];

  const int b   = blockIdx.x;
  const int tid = threadIdx.x;

  for (int i = tid; i < Q_ * K_; i += 256) ((float*)pool)[i] = 0.f;
  __syncthreads();

  const int wave  = tid >> 5;
  const int lane  = tid & 31;
  const int h     = lane >> 4;   // K-half selector for A/B fragments
  const int ln    = lane & 15;   // row (A) / column (B) within 16-tile
  const int mtile = wave & 1;
  const int ntres = wave >> 1;
  const int dlen  = dlen_arr[b];

  // Packed accumulators: acc2[rp][k] holds rows (2rp, 2rp+1).
  v2f acc2[4][K_];
#pragma unroll
  for (int rp = 0; rp < 4; ++rp)
#pragma unroll
    for (int k = 0; k < K_; ++k) acc2[rp][k] = (v2f){0.f, 0.f};

  // ---- Preload all 10 A fragments for this wave's mtile (kept in VGPRs) ---
  // A 16x32 f16 layout: lane row m = ln; K chunks [8h,8h+8) and [16+8h,+8).
  const _Float16* arow = qws + ((size_t)b * Q_ + (size_t)mtile * 16 + ln) * EP_;
  v16h afrag[10];
#pragma unroll
  for (int ks = 0; ks < 10; ++ks) {
    const int kb = ks * 32;
    v8h a0 = *(const v8h*)(arow + kb + 8 * h);
    v8h a1 = *(const v8h*)(arow + kb + 16 + 8 * h);
    afrag[ks] = cat16(a0, a1);
  }

  for (int j = 0; j < 8; ++j) {
    const int nt = ntres + 4 * j;
    const int n  = nt * 16 + ln;                 // doc column for this lane
    const _Float16* brow = dws + ((size_t)b * D_ + n) * EP_;

    // B 32x16 f16 layout: lane col n = ln; K chunks [8h,8h+8) and [16+8h,+8).
    v8h bf[20];
#pragma unroll
    for (int ks = 0; ks < 10; ++ks) {
      const int kb = ks * 32;
      bf[2 * ks]     = *(const v8h*)(brow + kb + 8 * h);
      bf[2 * ks + 1] = *(const v8h*)(brow + kb + 16 + 8 * h);
    }

    // Prefetch next tile's B rows (zero VGPR cost; hides next miss under
    // this tile's WMMA + pooling work).
    if (j < 7) {
      const _Float16* bnext = dws + ((size_t)b * D_ + (nt + 4) * 16 + ln) * EP_;
      __builtin_prefetch(bnext + h * 128, 0, 1);        // bytes h*256
      __builtin_prefetch(bnext + h * 128 + 64, 0, 1);   // bytes h*256+128
    }

    // ---- Dual-accumulator WMMA chain (break C RAW serialization) ----
    v8f c0 = {}, c1 = {};
#pragma unroll
    for (int ks = 0; ks < 10; ks += 2) {
      c0 = __builtin_amdgcn_wmma_f32_16x16x32_f16(
              false, afrag[ks],     false, cat16(bf[2*ks],   bf[2*ks+1]),
              (short)0, c0, false, false);
      c1 = __builtin_amdgcn_wmma_f32_16x16x32_f16(
              false, afrag[ks + 1], false, cat16(bf[2*ks+2], bf[2*ks+3]),
              (short)0, c1, false, false);
    }

    // ---- Fused Gaussian kernel pooling (geometric recurrence) ----
    // C/D layout: element r of C holds (q = mtile*16 + 8h + r, n).
    // term_k = exp(-50 (s-mu_k)^2); term_{k+1} = term_k * r * EK[k-1],
    // r = exp(-10 s). Only 3 transcendentals per sim value.
    if (n < dlen) {
#pragma unroll
      for (int rp = 0; rp < 4; ++rp) {
        const float s0 = c0[2 * rp]     + c1[2 * rp];
        const float s1 = c0[2 * rp + 1] + c1[2 * rp + 1];
        const float d00 = s0 - 1.0f,  d01 = s1 - 1.0f;   // k=0 (sigma 1e-3)
        const float d10 = s0 - 0.95f, d11 = s1 - 0.95f;  // k=1
        v2f p0 = { __builtin_amdgcn_exp2f(CC0 * d00 * d00),
                   __builtin_amdgcn_exp2f(CC0 * d01 * d01) };
        v2f t  = { __builtin_amdgcn_exp2f(CC1 * d10 * d10),
                   __builtin_amdgcn_exp2f(CC1 * d11 * d11) };
        v2f rr = { __builtin_amdgcn_exp2f(RC * s0),
                   __builtin_amdgcn_exp2f(RC * s1) };
        acc2[rp][0] += p0;
        acc2[rp][1] += t;
#pragma unroll
        for (int k = 2; k < K_; ++k) {
          t = t * rr;
          t = t * EK[k - 2];
          acc2[rp][k] += t;
        }
      }
    }
  }

  // Flush register accumulators into the shared pooling table.
#pragma unroll
  for (int rp = 0; rp < 4; ++rp) {
#pragma unroll
    for (int e = 0; e < 2; ++e) {
      const int q = mtile * 16 + 8 * h + 2 * rp + e;
#pragma unroll
      for (int k = 0; k < K_; ++k)
        atomicAdd(&pool[q][k], acc2[rp][k][e]);
    }
  }
  __syncthreads();

  // log_pooling_sum[b, k] = sum_q log(clip(pool)) * mask_q * 0.01
  const int qlen = qlen_arr[b];
  if (tid < K_) {
    float lsum = 0.f;
    for (int q = 0; q < Q_; ++q) {
      const float p  = fmaxf(pool[q][tid], 1e-10f);
      const float lg = __builtin_amdgcn_logf(p) * 0.69314718056f; // ln
      lsum += (q < qlen) ? lg * 0.01f : 0.f;
    }
    out[B_ + b * K_ + tid] = lsum;   // second tuple output, flat after score
    lps_sh[tid] = lsum;
  }
  __syncthreads();

  if (tid == 0) {
    float sc = dense_b[0];
#pragma unroll
    for (int k = 0; k < K_; ++k) sc += lps_sh[k] * dense_w[k];
    out[b] = sc;                     // first tuple output: score[b]
  }
}

// ---------------- launch ----------------
extern "C" void kernel_launch(void* const* d_in, const int* in_sizes, int n_in,
                              void* d_out, int out_size, void* d_ws, size_t ws_size,
                              hipStream_t stream) {
  const int*   qidx = (const int*)d_in[0];
  const int*   didx = (const int*)d_in[1];
  const int*   qlen = (const int*)d_in[2];
  const int*   dlen = (const int*)d_in[3];
  const float* emb  = (const float*)d_in[4];
  const float* w    = (const float*)d_in[5];
  const float* bias = (const float*)d_in[6];
  float* out = (float*)d_out;

  _Float16* qws = (_Float16*)d_ws;                       // B*Q*320 halfs
  _Float16* dws = qws + (size_t)B_ * Q_ * EP_;           // B*D*320 halfs
  // total ws use: (B*Q + B*D) * 320 * 2 bytes ~= 44.6 MB

  const int rows    = B_ * (Q_ + D_);                    // 69632 waves
  const int blocks1 = rows * 32 / 256;                   // 8704 blocks
  gather_norm_kernel<<<blocks1, 256, 0, stream>>>(qidx, didx, emb, qws, dws);
  knrm_pool_kernel<<<B_, 256, 0, stream>>>(qws, dws, qlen, dlen, w, bias, out);
}
